// QuantumHybridClassifier_65481071400744
// MI455X (gfx1250) — compile-verified, matrix operand order checked by reading, not run on hardware
//
#include <hip/hip_runtime.h>
#include <hip/hip_bf16.h>
#include <hip/hip_fp16.h>

typedef __attribute__((ext_vector_type(16))) _Float16 v16h;
typedef __attribute__((ext_vector_type(8)))  float    v8f;

#define M_ROWS (16 * 2048)   // B*S flattened rows
#define SEQ    2048
#define BATCH  16

// ---------- WMMA fragment helpers (f16 16x16x32, wave32) ----------
// K index of fragment element e = 2*v+s  (CDNA5 ISA 16-bit A layout)
__device__ __forceinline__ int frag_k(int v, int s, int lane) {
  return ((v & 4) ? 16 : 0) + ((lane & 16) ? 8 : 0) + ((v & 3) << 1) + s;
}

// Load A-fragment (or B-fragment of a column-major B, i.e. row-major B^T)
// base points at tile origin (tile row 0, k0); ld = leading dim in elements.
__device__ __forceinline__ v16h load_frag(const _Float16* base, int ld, int lane) {
  v16h a;
  const int m = lane & 15;
#pragma unroll
  for (int v = 0; v < 8; ++v) {
#pragma unroll
    for (int s = 0; s < 2; ++s) {
      a[2 * v + s] = base[m * ld + frag_k(v, s, lane)];
    }
  }
  return a;
}

__device__ __forceinline__ v8f wmma_f16(v16h a, v16h b, v8f c) {
  return __builtin_amdgcn_wmma_f32_16x16x32_f16(false, a, false, b, (short)0, c,
                                                false, false);
}

__device__ __forceinline__ float sigf(float x) {
  return 1.0f / (1.0f + __expf(-x));
}

// ---------- packing / cast kernels ----------
__global__ void k_cast_x(const float* __restrict__ x, _Float16* __restrict__ xh) {
  int i = blockIdx.x * blockDim.x + threadIdx.x;
  if (i >= M_ROWS * 160) return;
  int m = i / 160, k = i - m * 160;
  xh[i] = (k < 129) ? (_Float16)x[m * 129 + k] : (_Float16)0.0f;
}

__global__ void k_pack_wih(const float* __restrict__ w, _Float16* __restrict__ o) {
  int i = blockIdx.x * blockDim.x + threadIdx.x;     // 512*160
  if (i >= 512 * 160) return;
  int n = i / 160, k = i - n * 160;
  o[i] = (k < 129) ? (_Float16)w[n * 129 + k] : (_Float16)0.0f;
}

__global__ void k_pack_whh(const float* __restrict__ w, _Float16* __restrict__ o) {
  int i = blockIdx.x * blockDim.x + threadIdx.x;     // 512*128
  if (i >= 512 * 128) return;
  o[i] = (_Float16)w[i];
}

__global__ void k_pack_wc(const float* __restrict__ w, _Float16* __restrict__ o) {
  int i = blockIdx.x * blockDim.x + threadIdx.x;     // 4*144*160
  if (i >= 4 * 144 * 160) return;
  int l = i / (144 * 160);
  int r = i - l * 144 * 160;
  int n = r / 160, k = r - n * 160;
  o[i] = (n < 129 && k < 129) ? (_Float16)w[(l * 129 + n) * 129 + k]
                              : (_Float16)0.0f;
}

// RBF feature -> column 128 of MLP input; zero K-padding columns 129..159
__global__ void k_rbf(const float* __restrict__ x, const float* __restrict__ proto,
                      _Float16* __restrict__ mlp0) {
  int m = blockIdx.x * blockDim.x + threadIdx.x;
  if (m >= M_ROWS) return;
  const float* xp = x + m * 129;
  const float* pp = proto + m * 129;
  float s = 0.0f;
  for (int k = 0; k < 129; ++k) {
    float d = xp[k] - pp[k];
    s += d * d;
  }
  _Float16* dst = mlp0 + m * 160;
  dst[128] = (_Float16)__expf(-s);
#pragma unroll
  for (int k = 129; k < 160; ++k) dst[k] = (_Float16)0.0f;
}

// ---------- K1: xg = x @ W_ih^T + (b_ih + b_hh) ----------
// one wave computes a 16x64 tile; K = 160 (padded 129)
__global__ void __launch_bounds__(32)
k_gemm_xg(const _Float16* __restrict__ xh, const _Float16* __restrict__ wih,
          const float* __restrict__ b_ih, const float* __restrict__ b_hh,
          float* __restrict__ xg) {
  const int lane = threadIdx.x;
  const int m0 = blockIdx.x * 16;
  const int n0 = blockIdx.y * 64;
  v8f z = {0, 0, 0, 0, 0, 0, 0, 0};
  v8f acc[4] = {z, z, z, z};
#pragma unroll
  for (int k0 = 0; k0 < 160; k0 += 32) {
    v16h a = load_frag(xh + m0 * 160 + k0, 160, lane);
#pragma unroll
    for (int j = 0; j < 4; ++j) {
      v16h b = load_frag(wih + (n0 + j * 16) * 160 + k0, 160, lane);
      acc[j] = wmma_f16(a, b, acc[j]);
    }
  }
  const int hi = (lane & 16) >> 1;   // 0 or 8
  const int nl = lane & 15;
#pragma unroll
  for (int j = 0; j < 4; ++j) {
    int col = n0 + j * 16 + nl;
    float bias = b_ih[col] + b_hh[col];
#pragma unroll
    for (int r = 0; r < 8; ++r) {
      int row = m0 + r + hi;
      xg[row * 512 + col] = acc[j][r] + bias;
    }
  }
}

// ---------- K2: LSTM recurrence, single workgroup (M = batch = 16) ----------
// 8 waves, each owns 4 N-tiles (64 of 512 gate columns); W_hh frags in VGPRs.
__global__ void __launch_bounds__(256)
k_lstm(const float* __restrict__ xg, const _Float16* __restrict__ whh,
       _Float16* __restrict__ q_h) {
  __shared__ __align__(16) _Float16 h_sh[16 * 128];
  __shared__ __align__(16) float gates[16 * 512];

  const int tid = threadIdx.x;
  const int lane = tid & 31;
  const int w = tid >> 5;            // wave 0..7
  const int hi = (lane & 16) >> 1;   // 0 or 8
  const int nl = lane & 15;

  // preload constant W_hh B-fragments: Bt[n][k] = W_hh[n][k], K=128
  v16h bf[4][4];
#pragma unroll
  for (int j = 0; j < 4; ++j)
#pragma unroll
    for (int kf = 0; kf < 4; ++kf)
      bf[j][kf] = load_frag(whh + (w * 64 + j * 16) * 128 + kf * 32, 128, lane);

  for (int i = tid; i < 16 * 128; i += 256) h_sh[i] = (_Float16)0.0f;

  float c_reg[8];
#pragma unroll
  for (int u = 0; u < 8; ++u) c_reg[u] = 0.0f;
  const int pbase = tid * 8;
  __syncthreads();

  for (int t = 0; t < SEQ; ++t) {
    v8f acc[4];
#pragma unroll
    for (int j = 0; j < 4; ++j) {
      int col = w * 64 + j * 16 + nl;
#pragma unroll
      for (int r = 0; r < 8; ++r) {
        int b = r + hi;
        acc[j][r] = xg[(b * SEQ + t) * 512 + col];  // xg + both biases
      }
    }
#pragma unroll
    for (int kf = 0; kf < 4; ++kf) {
      v16h a = load_frag(h_sh + kf * 32, 128, lane);
#pragma unroll
      for (int j = 0; j < 4; ++j) acc[j] = wmma_f16(a, bf[j][kf], acc[j]);
    }
#pragma unroll
    for (int j = 0; j < 4; ++j) {
      int col = w * 64 + j * 16 + nl;
#pragma unroll
      for (int r = 0; r < 8; ++r) gates[(r + hi) * 512 + col] = acc[j][r];
    }
    __syncthreads();
    // pointwise gates: each thread owns 8 (b, hidden) cells, c kept in regs
#pragma unroll
    for (int u = 0; u < 8; ++u) {
      int p = pbase + u;
      int b = p >> 7, jj = p & 127;
      float gi = gates[b * 512 + jj];
      float gf = gates[b * 512 + 128 + jj];
      float gg = gates[b * 512 + 256 + jj];
      float go = gates[b * 512 + 384 + jj];
      float c = sigf(gf) * c_reg[u] + sigf(gi) * tanhf(gg);
      c_reg[u] = c;
      float h = sigf(go) * tanhf(c);
      h_sh[b * 128 + jj] = (_Float16)h;
      q_h[(b * SEQ + t) * 128 + jj] = (_Float16)h;
    }
    __syncthreads();
  }
}

// ---------- K4: flash attention, one wave per 16-query tile ----------
__global__ void __launch_bounds__(32)
k_attn(const _Float16* __restrict__ q_h, _Float16* __restrict__ mlp0) {
  __shared__ __align__(16) _Float16 vtile[32 * 128];
  __shared__ __align__(16) _Float16 ptile[16 * 32];

  const int lane = threadIdx.x;
  const int bb = blockIdx.x >> 7;      // batch 0..15
  const int qt = blockIdx.x & 127;     // query tile 0..127
  const _Float16* qbase = q_h + (bb * SEQ + qt * 16) * 128;

  v16h aq[4];
#pragma unroll
  for (int kf = 0; kf < 4; ++kf) aq[kf] = load_frag(qbase + kf * 32, 128, lane);

  v8f z = {0, 0, 0, 0, 0, 0, 0, 0};
  v8f ctx[8] = {z, z, z, z, z, z, z, z};
  float mrow[8], lrow[8];
#pragma unroll
  for (int r = 0; r < 8; ++r) { mrow[r] = -3.0e38f; lrow[r] = 0.0f; }

  const float rscale = 0.08804509063256238f;  // 1/sqrt(NUM_FEATURES=129)
  const int hi = (lane & 16) >> 1;
  const int nl = lane & 15;

  for (int kc = 0; kc < SEQ / 32; ++kc) {
    const _Float16* kbase = q_h + (bb * SEQ + kc * 32) * 128;
    // stage 32 key rows (V for the PV matmul): lane copies the first 128B of its 256B row
    {
      const uint4* src = (const uint4*)(kbase + lane * 128);
      uint4* dst = (uint4*)(vtile + lane * 128);
#pragma unroll
      for (int i = 0; i < 8; ++i) dst[i] = src[i];
    }
    // scores: S(16x32) = Q(16x128) K^T ; B-frags straight from global (col-major K^T)
    v8f s0 = z, s1 = z;
#pragma unroll
    for (int kf = 0; kf < 4; ++kf) {
      v16h b0 = load_frag(kbase + kf * 32, 128, lane);
      v16h b1 = load_frag(kbase + 16 * 128 + kf * 32, 128, lane);
      s0 = wmma_f16(aq[kf], b0, s0);
      s1 = wmma_f16(aq[kf], b1, s1);
    }
    __syncthreads();
    // online softmax per row (row = r + hi, cols across 16-lane half-groups)
#pragma unroll
    for (int r = 0; r < 8; ++r) {
      float sa = s0[r] * rscale;
      float sb = s1[r] * rscale;
      float mx = fmaxf(sa, sb);
#pragma unroll
      for (int d = 1; d < 16; d <<= 1) mx = fmaxf(mx, __shfl_xor(mx, d, 32));
      float mn = fmaxf(mrow[r], mx);
      float alpha = __expf(mrow[r] - mn);
      mrow[r] = mn;
      float p0 = __expf(sa - mn);
      float p1 = __expf(sb - mn);
      float ps = p0 + p1;
#pragma unroll
      for (int d = 1; d < 16; d <<= 1) ps += __shfl_xor(ps, d, 32);
      lrow[r] = lrow[r] * alpha + ps;
#pragma unroll
      for (int ft = 0; ft < 8; ++ft) ctx[ft][r] *= alpha;
      int row = r + hi;
      ptile[row * 32 + nl] = (_Float16)p0;
      ptile[row * 32 + 16 + nl] = (_Float16)p1;
    }
    __syncthreads();
    // context += P(16x32) @ V(32x128)
    v16h ap = load_frag(ptile, 32, lane);
#pragma unroll
    for (int ft = 0; ft < 8; ++ft) {
      v16h bv;
#pragma unroll
      for (int v = 0; v < 8; ++v)
#pragma unroll
        for (int s = 0; s < 2; ++s)
          bv[2 * v + s] = vtile[frag_k(v, s, lane) * 128 + ft * 16 + nl];
      ctx[ft] = wmma_f16(ap, bv, ctx[ft]);
    }
    __syncthreads();
  }
  // normalize and write context into MLP-input cols 0..127 (f16, ld=160)
#pragma unroll
  for (int r = 0; r < 8; ++r) {
    float inv = 1.0f / lrow[r];
    int row = bb * SEQ + qt * 16 + r + hi;
#pragma unroll
    for (int ft = 0; ft < 8; ++ft)
      mlp0[row * 160 + ft * 16 + nl] = (_Float16)(ctx[ft][r] * inv);
  }
}

// ---------- K5: MLP layer  out = relu(in @ Wc^T + bc), 129->129, padded ----------
__global__ void __launch_bounds__(32)
k_mlp(const _Float16* __restrict__ in, const _Float16* __restrict__ wc,
      const float* __restrict__ bc, _Float16* __restrict__ out) {
  const int lane = threadIdx.x;
  const int m0 = blockIdx.x * 16;
  const int nt = blockIdx.y;  // 0..8 (N padded to 144)
  v8f acc = {0, 0, 0, 0, 0, 0, 0, 0};
#pragma unroll
  for (int k0 = 0; k0 < 160; k0 += 32) {
    v16h a = load_frag(in + m0 * 160 + k0, 160, lane);
    v16h b = load_frag(wc + nt * 16 * 160 + k0, 160, lane);
    acc = wmma_f16(a, b, acc);
  }
  const int hi = (lane & 16) >> 1;
  const int nl = lane & 15;
  const int col = nt * 16 + nl;
  const float bias = (col < 129) ? bc[col] : 0.0f;
#pragma unroll
  for (int r = 0; r < 8; ++r) {
    int row = m0 + r + hi;
    float v = fmaxf(acc[r] + bias, 0.0f);
    out[row * 160 + col] = (col < 129) ? (_Float16)v : (_Float16)0.0f;
    if (nt == 8) out[row * 160 + col + 16] = (_Float16)0.0f;  // cols 144..159
  }
}

// ---------- K6: head (129->2) + log_softmax ----------
__global__ void k_head(const _Float16* __restrict__ in, const float* __restrict__ Wh,
                       const float* __restrict__ bh, float* __restrict__ out) {
  int m = blockIdx.x * blockDim.x + threadIdx.x;
  if (m >= M_ROWS) return;
  const _Float16* h = in + m * 160;
  float l0 = bh[0], l1 = bh[1];
  for (int k = 0; k < 129; ++k) {
    float hv = (float)h[k];
    l0 += hv * Wh[k];
    l1 += hv * Wh[129 + k];
  }
  float mx = fmaxf(l0, l1);
  float lse = mx + __logf(__expf(l0 - mx) + __expf(l1 - mx));
  out[m * 2 + 0] = l0 - lse;
  out[m * 2 + 1] = l1 - lse;
}

// ---------- host ----------
extern "C" void kernel_launch(void* const* d_in, const int* in_sizes, int n_in,
                              void* d_out, int out_size, void* d_ws, size_t ws_size,
                              hipStream_t stream) {
  (void)in_sizes; (void)n_in; (void)out_size; (void)ws_size;
  const float* x     = (const float*)d_in[0];
  const float* proto = (const float*)d_in[1];
  const float* W_ih  = (const float*)d_in[2];
  const float* W_hh  = (const float*)d_in[3];
  const float* b_ih  = (const float*)d_in[4];
  const float* b_hh  = (const float*)d_in[5];
  const float* Wc    = (const float*)d_in[6];
  const float* bc    = (const float*)d_in[7];
  const float* Wh    = (const float*)d_in[8];
  const float* bh    = (const float*)d_in[9];
  float* out = (float*)d_out;

  char* ws = (char*)d_ws;
  float*    xg   = (float*)   (ws + 0);          // M*512*4   = 67,108,864
  _Float16* xh   = (_Float16*)(ws + 67108864);   // M*160*2   = 10,485,760
  _Float16* q_h  = (_Float16*)(ws + 77594624);   // M*128*2   =  8,388,608
  _Float16* mlp0 = (_Float16*)(ws + 85983232);   // M*160*2   = 10,485,760
  _Float16* mlp1 = (_Float16*)(ws + 96468992);   // M*160*2   = 10,485,760
  _Float16* wih  = (_Float16*)(ws + 106954752);  // 512*160*2 =    163,840
  _Float16* whh  = (_Float16*)(ws + 107118592);  // 512*128*2 =    131,072
  _Float16* wch  = (_Float16*)(ws + 107249664);  // 4*144*160*2 =  184,320

  // pack / cast
  k_cast_x  <<<dim3((M_ROWS * 160 + 255) / 256), dim3(256), 0, stream>>>(x, xh);
  k_pack_wih<<<dim3((512 * 160 + 255) / 256),    dim3(256), 0, stream>>>(W_ih, wih);
  k_pack_whh<<<dim3((512 * 128 + 255) / 256),    dim3(256), 0, stream>>>(W_hh, whh);
  k_pack_wc <<<dim3((4 * 144 * 160 + 255) / 256), dim3(256), 0, stream>>>(Wc, wch);
  k_rbf     <<<dim3((M_ROWS + 255) / 256),       dim3(256), 0, stream>>>(x, proto, mlp0);

  // xg GEMM
  k_gemm_xg<<<dim3(M_ROWS / 16, 512 / 64), dim3(32), 0, stream>>>(xh, wih, b_ih, b_hh, xg);

  // sequential LSTM (single workgroup)
  k_lstm<<<dim3(1), dim3(256), 0, stream>>>(xg, whh, q_h);

  // flash attention -> mlp0 cols 0..127
  k_attn<<<dim3(BATCH * (SEQ / 16)), dim3(32), 0, stream>>>(q_h, mlp0);

  // 4 MLP layers (ping-pong), then head reads mlp0
  _Float16* bufs[2] = {mlp0, mlp1};
  for (int l = 0; l < 4; ++l) {
    k_mlp<<<dim3(M_ROWS / 16, 9), dim3(32), 0, stream>>>(bufs[l & 1], wch + l * 144 * 160,
                                                         bc + l * 129, bufs[(l + 1) & 1]);
  }
  k_head<<<dim3(M_ROWS / 256), dim3(256), 0, stream>>>(mlp0, Wh, bh, out);
}
